// LeViTAttention_75239237092065
// MI455X (gfx1250) — matrix operand order repaired and checked by reading, not compiled, not run on hardware
//
#include <hip/hip_runtime.h>
#include <hip/hip_bf16.h>

// MI455X / gfx1250 LeViT-attention pipeline.
// All matrix math on v_wmma_f32_16x16x32_f16 (K=32 == head dim).
// All WMMA operands but V (per-element gather) are fed by 16-byte global_load_b128 / ds_load_b128:
//  - A layout (16x32 f16): per lane one contiguous 16-half run.
//  - B layout (32x16 f16): per lane one contiguous 16-half run, provided the
//    source is stored K-contiguous (weights pre-transposed, x row-major).
//
// ws layout:
//   xh  : [25088][1024] f16   qkv output; unnormalized, then BN'd in place
//   ah  : [25088][256]  f16   packed inputs
//   wqh : [1024][256]   f16   w_qkv transposed (n-major, k-contiguous)
//   woh : [256][512]    f16   w_out transposed
//   tmp : [25088][512]  f16   hard_swish(attn@v), head-major
//   ps  : [1568][1024]  f32   per-row-tile column sums   (pre-f16-rounding)
//   pq  : [1568][1024]  f32   per-row-tile column sumsq
//   alpha,beta : [1024] f32   folded BN affine
// total ~110 MB.

typedef __attribute__((ext_vector_type(16))) _Float16 v16h;
typedef __attribute__((ext_vector_type(8)))  _Float16 v8h;
typedef __attribute__((ext_vector_type(4)))  _Float16 v4h;
typedef __attribute__((ext_vector_type(8)))  float    v8f;

#define WMMA_F16(A, B, C) \
  __builtin_amdgcn_wmma_f32_16x16x32_f16(false, (A), false, (B), (short)0, (C), false, false)

#define CAT16(LO, HI) \
  __builtin_shufflevector((LO), (HI), 0, 1, 2, 3, 4, 5, 6, 7, 8, 9, 10, 11, 12, 13, 14, 15)

namespace {
constexpr int kC     = 256;    // input channels
constexpr int kF     = 1024;   // qkv features
constexpr int kN     = 784;    // tokens per image
constexpr int kM     = 25088;  // B * N
constexpr int kMT    = 1568;   // kM / 16 row tiles
constexpr int kHeads = 8;
}

// ---------------------------------------------------------------------------
// Packing kernels: fp32 -> f16 (weights transposed so B operands are
// K-contiguous => b128 loads in the GEMM inner loops).
// ---------------------------------------------------------------------------
__global__ void __launch_bounds__(256) pack_in(const float* __restrict__ in,
                                               _Float16* __restrict__ ah) {
  const size_t i = ((size_t)blockIdx.x * 256 + threadIdx.x) * 4;  // 6272 blocks
  const float4 v = *(const float4*)(in + i);
  v4h o;
  o[0] = (_Float16)v.x; o[1] = (_Float16)v.y;
  o[2] = (_Float16)v.z; o[3] = (_Float16)v.w;
  *(v4h*)(ah + i) = o;
}

__global__ void __launch_bounds__(256) pack_wq(const float* __restrict__ w,
                                               _Float16* __restrict__ wh) {
  const int idx = blockIdx.x * 256 + threadIdx.x;  // [0, 1024*256)
  const int n = idx >> 8;
  const int k = idx & 255;
  wh[idx] = (_Float16)w[(size_t)k * kF + n];       // wh[n][k] = w[k][n]
}

__global__ void __launch_bounds__(256) pack_wo(const float* __restrict__ w,
                                               _Float16* __restrict__ wh) {
  const int idx = blockIdx.x * 256 + threadIdx.x;  // [0, 256*512)
  const int n = idx >> 9;
  const int k = idx & 511;
  wh[idx] = (_Float16)w[(size_t)k * kC + n];       // wh[n][k] = w[k][n]
}

// ---------------------------------------------------------------------------
// Kernel 1: xh[25088,1024] = ah[25088,256] @ wqh^T ; also emits per-tile
// f32 column sums / sumsq for deterministic BN stats.
// one wave per 16x16 tile; 1568*64 tiles; 8 waves/block -> 12544 blocks.
// ---------------------------------------------------------------------------
__global__ void __launch_bounds__(256) qkv_gemm(const _Float16* __restrict__ ah,
                                                const _Float16* __restrict__ wqh,
                                                _Float16* __restrict__ xh,
                                                float* __restrict__ ps,
                                                float* __restrict__ pq) {
  const int wave = threadIdx.x >> 5;
  const int lane = threadIdx.x & 31;
  const int tile = blockIdx.x * 8 + wave;   // [0, 1568*64)
  const int mt   = tile >> 6;
  const int ft   = tile & 63;
  const int m    = lane & 15;
  const int hi   = lane >> 4;

  const _Float16* arow = ah  + (size_t)(mt * 16 + m) * kC;
  const _Float16* brow = wqh + (size_t)(ft * 16 + m) * kC;
  v8f acc = {};
  for (int k0 = 0; k0 < kC; k0 += 32) {
    if (k0 + 32 < kC) __builtin_prefetch(arow + k0 + 32, 0, 0);
    const v8h* ap = (const v8h*)(arow + k0 + hi * 16);  // A: K = hi*16+e
    const v8h* bp = (const v8h*)(brow + k0 + hi * 16);  // B: K = hi*16+e
    v16h a = CAT16(ap[0], ap[1]);
    v16h b = CAT16(bp[0], bp[1]);
    acc = WMMA_F16(a, b, acc);
  }
  // store unnormalized f16 (C layout: vgpr r -> row r+8*hi, col m)
  _Float16* cp = xh + (size_t)(mt * 16 + hi * 8) * kF + ft * 16 + m;
#pragma unroll
  for (int r = 0; r < 8; ++r) cp[(size_t)r * kF] = (_Float16)acc[r];
  // column partial sums over the tile's 16 rows (f32, pre-rounding)
  float s = 0.f, q = 0.f;
#pragma unroll
  for (int r = 0; r < 8; ++r) { s += acc[r]; q += acc[r] * acc[r]; }
  s += __shfl_xor(s, 16, 32);
  q += __shfl_xor(q, 16, 32);
  if (hi == 0) {
    ps[(size_t)mt * kF + ft * 16 + m] = s;
    pq[(size_t)mt * kF + ft * 16 + m] = q;
  }
}

// ---------------------------------------------------------------------------
// Kernel 2: reduce 1568 tile partials per channel -> alpha/beta. Deterministic.
// ---------------------------------------------------------------------------
__global__ void __launch_bounds__(256) bn_stats(const float* __restrict__ ps,
                                                const float* __restrict__ pq,
                                                const float* __restrict__ scale,
                                                float* __restrict__ alpha,
                                                float* __restrict__ beta) {
  __shared__ float ssum[256];
  __shared__ float ssq[256];
  const int f = blockIdx.x;
  float s = 0.f, q = 0.f;
  for (int mt = threadIdx.x; mt < kMT; mt += 256) {
    s += ps[(size_t)mt * kF + f];
    q += pq[(size_t)mt * kF + f];
  }
  ssum[threadIdx.x] = s;
  ssq[threadIdx.x]  = q;
  __syncthreads();
  for (int off = 128; off > 0; off >>= 1) {
    if ((int)threadIdx.x < off) {
      ssum[threadIdx.x] += ssum[threadIdx.x + off];
      ssq[threadIdx.x]  += ssq[threadIdx.x + off];
    }
    __syncthreads();
  }
  if (threadIdx.x == 0) {
    float mean = ssum[0] * (1.0f / kM);
    float var  = ssq[0] * (1.0f / kM) - mean * mean;
    float a    = scale[f] * rsqrtf(var + 1e-5f);
    alpha[f] = a;
    beta[f]  = -mean * a;
  }
}

// ---------------------------------------------------------------------------
// Kernel 2b: BN affine applied to xh in place (8 halves per thread).
// ---------------------------------------------------------------------------
__global__ void __launch_bounds__(256) bn_apply(_Float16* __restrict__ xh,
                                                const float* __restrict__ alpha,
                                                const float* __restrict__ beta) {
  const size_t i = ((size_t)blockIdx.x * 256 + threadIdx.x) * 8;  // 12544 blocks
  const int f0 = (int)(i & (size_t)(kF - 1));
  v8h v = *(const v8h*)(xh + i);
  v8h o;
#pragma unroll
  for (int e = 0; e < 8; ++e)
    o[e] = (_Float16)((float)v[e] * alpha[f0 + e] + beta[f0 + e]);
  *(v8h*)(xh + i) = o;
}

// ---------------------------------------------------------------------------
// Kernel 3: flash attention per (batch, head, 16-query tile).
// one wave per tile; 32*8*49 = 12544 tiles; 4 waves/block -> 3136 blocks.
// Key chunks of 32 (25 chunks; keys >= 784 masked with -1e30 scores).
// ---------------------------------------------------------------------------
__global__ void __launch_bounds__(128) attn_kernel(const _Float16* __restrict__ xh,
                                                   const float* __restrict__ bias,
                                                   _Float16* __restrict__ tmp) {
  __shared__ _Float16 lp[4][16 * 32];   // per-wave P scratch (C-layout -> A-layout)
  const int wave = threadIdx.x >> 5;
  const int lane = threadIdx.x & 31;
  const int t    = blockIdx.x * 4 + wave;        // [0, 32*8*49)
  const int b    = t / (kHeads * 49);
  const int rem  = t % (kHeads * 49);
  const int h    = rem / 49;
  const int qt   = rem % 49;
  const int m    = lane & 15;
  const int hi   = lane >> 4;
  const int qrow = qt * 16;

  // Q tile in A layout: one contiguous 16-half run -> 2x b128
  const v8h* qp =
      (const v8h*)(xh + (size_t)(b * kN + qrow + m) * kF + h * 32 + hi * 16);
  v16h qa = CAT16(qp[0], qp[1]);

  // V channel per (d, lane): dim = d*16 + m
  int vch[4];
#pragma unroll
  for (int d = 0; d < 4; ++d) {
    const int dim = d * 16 + m;
    vch[d] = (dim < 32) ? (512 + h * 32 + dim) : (768 + h * 32 + (dim - 32));
  }

  float mi[8], li[8];
  v8f o[4] = {};
#pragma unroll
  for (int r = 0; r < 8; ++r) { mi[r] = -__builtin_inff(); li[r] = 0.f; }

  const size_t xb    = (size_t)b * kN * kF;
  const int    kchan = 256 + h * 32 + hi * 16;   // K features, contiguous run

  for (int jc = 0; jc < kN; jc += 32) {
    const int key0 = jc + m;
    const int key1 = jc + 16 + m;

    // ---- scores: C initialized with attn_bias, D = Q*K^T + bias ----
    v8f c0, c1;
    {
      const size_t br = (size_t)(qrow + hi * 8) * kN;
#pragma unroll
      for (int r = 0; r < 8; ++r) {
        c0[r] = (key0 < kN) ? bias[br + (size_t)r * kN + key0] : -1e30f;
        c1[r] = (key1 < kN) ? bias[br + (size_t)r * kN + key1] : -1e30f;
      }
    }
    v16h kb0 = {}, kb1 = {};
    if (key0 < kN) {
      const v8h* kp = (const v8h*)(xh + xb + (size_t)key0 * kF + kchan);
      kb0 = CAT16(kp[0], kp[1]);
    }
    if (key1 < kN) {
      const v8h* kp = (const v8h*)(xh + xb + (size_t)key1 * kF + kchan);
      kb1 = CAT16(kp[0], kp[1]);
    }
    c0 = WMMA_F16(qa, kb0, c0);
    c1 = WMMA_F16(qa, kb1, c1);

    // ---- online softmax; rows live in 16-lane groups -> xor-shuffle reduce ----
#pragma unroll
    for (int r = 0; r < 8; ++r) {
      float cm = fmaxf(c0[r], c1[r]);
#pragma unroll
      for (int s2 = 1; s2 < 16; s2 <<= 1) cm = fmaxf(cm, __shfl_xor(cm, s2, 32));
      const float mn   = fmaxf(mi[r], cm);
      const float corr = __expf(mi[r] - mn);
      mi[r] = mn;
      const float p0 = __expf(c0[r] - mn);
      const float p1 = __expf(c1[r] - mn);
      c0[r] = p0;
      c1[r] = p1;
      float pssum = p0 + p1;
#pragma unroll
      for (int s2 = 1; s2 < 16; s2 <<= 1) pssum += __shfl_xor(pssum, s2, 32);
      li[r] = li[r] * corr + pssum;
#pragma unroll
      for (int d = 0; d < 4; ++d) o[d][r] = o[d][r] * corr;
    }

    // ---- P: C layout -> f16 A layout via wave-private LDS ----
    _Float16* w = lp[wave];
#pragma unroll
    for (int r = 0; r < 8; ++r) {
      w[(r + 8 * hi) * 32 + m]      = (_Float16)c0[r];
      w[(r + 8 * hi) * 32 + 16 + m] = (_Float16)c1[r];
    }
    asm volatile("s_wait_dscnt 0" ::: "memory");   // wave-local DS flush
    const v8h* pr = (const v8h*)(w + m * 32 + hi * 16);  // 2x ds_load_b128
    v16h pa = CAT16(pr[0], pr[1]);
    asm volatile("s_wait_dscnt 0" ::: "memory");

    // ---- o += P @ V ; V as four 32x16 B tiles (dims 0..63) ----
#pragma unroll
    for (int d = 0; d < 4; ++d) {
      v16h vb;
#pragma unroll
      for (int e = 0; e < 16; ++e) {
        const int key = jc + hi * 16 + e;
        vb[e] = (key < kN) ? xh[xb + (size_t)key * kF + vch[d]] : (_Float16)0.f;
      }
      o[d] = WMMA_F16(pa, vb, o[d]);
    }
  }

  // ---- epilogue: 1/l, hard_swish, store f16 [25088, 512] head-major ----
#pragma unroll
  for (int r = 0; r < 8; ++r) {
    const float inv  = 1.0f / li[r];
    const size_t orw = (size_t)(b * kN + qrow + r + 8 * hi) * 512 + h * 64;
#pragma unroll
    for (int d = 0; d < 4; ++d) {
      const float v  = o[d][r] * inv;
      const float hs = v * fminf(fmaxf(v + 3.f, 0.f), 6.f) * (1.f / 6.f);
      tmp[orw + d * 16 + m] = (_Float16)hs;
    }
  }
}

// ---------------------------------------------------------------------------
// Kernel 4: out[25088,256] = tmp[25088,512] @ woh^T
// 1568*16 tiles; 8 waves/block -> 3136 blocks.
// ---------------------------------------------------------------------------
__global__ void __launch_bounds__(256) out_gemm(const _Float16* __restrict__ tmp,
                                                const _Float16* __restrict__ woh,
                                                float* __restrict__ out) {
  const int wave = threadIdx.x >> 5;
  const int lane = threadIdx.x & 31;
  const int tile = blockIdx.x * 8 + wave;   // [0, 1568*16)
  const int mt   = tile >> 4;
  const int ft   = tile & 15;
  const int m    = lane & 15;
  const int hi   = lane >> 4;

  const _Float16* arow = tmp + (size_t)(mt * 16 + m) * 512;
  const _Float16* brow = woh + (size_t)(ft * 16 + m) * 512;
  v8f acc = {};
  for (int k0 = 0; k0 < 512; k0 += 32) {
    if (k0 + 32 < 512) __builtin_prefetch(arow + k0 + 32, 0, 0);
    const v8h* ap = (const v8h*)(arow + k0 + hi * 16);
    const v8h* bp = (const v8h*)(brow + k0 + hi * 16);
    v16h a = CAT16(ap[0], ap[1]);
    v16h b = CAT16(bp[0], bp[1]);
    acc = WMMA_F16(a, b, acc);
  }
  float* cp = out + (size_t)(mt * 16 + hi * 8) * kC + ft * 16 + m;
#pragma unroll
  for (int r = 0; r < 8; ++r) cp[(size_t)r * kC] = acc[r];
}

// ---------------------------------------------------------------------------
extern "C" void kernel_launch(void* const* d_in, const int* in_sizes, int n_in,
                              void* d_out, int out_size, void* d_ws, size_t ws_size,
                              hipStream_t stream) {
  (void)in_sizes; (void)n_in; (void)out_size; (void)ws_size;
  const float* inp  = (const float*)d_in[0];   // [32,28,28,256]
  const float* wqkv = (const float*)d_in[1];   // [256,1024]
  const float* bns  = (const float*)d_in[2];   // [1024]
  const float* bias = (const float*)d_in[3];   // [1,1,784,784]
  const float* wout = (const float*)d_in[4];   // [512,256]
  float* out = (float*)d_out;                  // [32,28,28,256]

  char* p = (char*)d_ws;
  _Float16* xh  = (_Float16*)p;  p += (size_t)kM * kF * 2;
  _Float16* ah  = (_Float16*)p;  p += (size_t)kM * kC * 2;
  _Float16* wqh = (_Float16*)p;  p += (size_t)kF * kC * 2;
  _Float16* woh = (_Float16*)p;  p += (size_t)256 * 512 * 2;
  _Float16* tmp = (_Float16*)p;  p += (size_t)kM * 512 * 2;
  float*    ps  = (float*)p;     p += (size_t)kMT * kF * 4;
  float*    pq  = (float*)p;     p += (size_t)kMT * kF * 4;
  float*    alpha = (float*)p;   p += (size_t)kF * 4;
  float*    beta  = (float*)p;

  pack_in<<<6272, 256, 0, stream>>>(inp, ah);        // 25088*256 / 4
  pack_wq<<<1024, 256, 0, stream>>>(wqkv, wqh);      // 1024*256
  pack_wo<<<512, 256, 0, stream>>>(wout, woh);       // 256*512
  qkv_gemm<<<12544, 256, 0, stream>>>(ah, wqh, xh, ps, pq);
  bn_stats<<<kF, 256, 0, stream>>>(ps, pq, bns, alpha, beta);
  bn_apply<<<12544, 256, 0, stream>>>(xh, alpha, beta);
  attn_kernel<<<3136, 128, 0, stream>>>(xh, bias, tmp);
  out_gemm<<<3136, 256, 0, stream>>>(tmp, woh, out);
}
